// MambaBlock_24824910970934
// MI455X (gfx1250) — matrix operand order repaired and checked by reading, not compiled, not run on hardware
//
#include <hip/hip_runtime.h>
#include <hip/hip_bf16.h>
#include <hip/hip_fp16.h>

#define D_MODEL 768
#define D_INNER 1536
#define D2      3072
#define D_STATE 16
#define DT_RANK 48
#define DT_PAD  64
#define NXPROJ  80      /* DT_RANK + 2*D_STATE */
#define NB      2
#define T_LEN   1024
#define M_TOK   2048    /* NB * T_LEN */

typedef _Float16 v16h __attribute__((ext_vector_type(16)));
typedef _Float16 v8h  __attribute__((ext_vector_type(8)));
typedef float    v8f  __attribute__((ext_vector_type(8)));
typedef int      v4i  __attribute__((ext_vector_type(4)));

union Frag16 { v16h v; v8h h[2]; };

#define AS1 __attribute__((address_space(1)))
#define AS3 __attribute__((address_space(3)))

#if defined(__AMDGCN__) && \
    __has_builtin(__builtin_amdgcn_global_load_async_to_lds_b128) && \
    __has_builtin(__builtin_amdgcn_s_wait_asynccnt)
#define USE_ASYNC 1
#else
#define USE_ASYNC 0
#endif

// ---------------------------------------------------------------------------
// LayerNorm over D_MODEL, output f16 for WMMA GEMM consumption
// ---------------------------------------------------------------------------
__global__ __launch_bounds__(256)
void ln_kernel(const float* __restrict__ x, const float* __restrict__ g,
               const float* __restrict__ b, _Float16* __restrict__ h) {
    const int tok = blockIdx.x;
    const float* xr = x + (size_t)tok * D_MODEL;
    __shared__ float red[256];
    const int tid = threadIdx.x;
    float v0 = xr[tid], v1 = xr[tid + 256], v2 = xr[tid + 512];
    red[tid] = v0 + v1 + v2;
    __syncthreads();
    for (int o = 128; o > 0; o >>= 1) {
        if (tid < o) red[tid] += red[tid + o];
        __syncthreads();
    }
    const float mu = red[0] * (1.0f / D_MODEL);
    __syncthreads();
    const float d0 = v0 - mu, d1 = v1 - mu, d2 = v2 - mu;
    red[tid] = d0 * d0 + d1 * d1 + d2 * d2;
    __syncthreads();
    for (int o = 128; o > 0; o >>= 1) {
        if (tid < o) red[tid] += red[tid + o];
        __syncthreads();
    }
    const float rs = rsqrtf(red[0] * (1.0f / D_MODEL) + 1e-5f);
    _Float16* hr = h + (size_t)tok * D_MODEL;
    hr[tid]       = (_Float16)(d0 * rs * g[tid]       + b[tid]);
    hr[tid + 256] = (_Float16)(d1 * rs * g[tid + 256] + b[tid + 256]);
    hr[tid + 512] = (_Float16)(d2 * rs * g[tid + 512] + b[tid + 512]);
}

// ---------------------------------------------------------------------------
// f32 -> f16 conversion / padding helpers
// ---------------------------------------------------------------------------
__global__ void cvt_kernel(const float* __restrict__ src, _Float16* __restrict__ dst, int n) {
    int i = blockIdx.x * 256 + threadIdx.x;
    if (i < n) dst[i] = (_Float16)src[i];
}

__global__ void pad_wdt_kernel(const float* __restrict__ src, _Float16* __restrict__ dst) {
    int i = blockIdx.x * 256 + threadIdx.x;
    if (i >= D_INNER * DT_PAD) return;
    int r = i / DT_PAD, c = i % DT_PAD;
    dst[i] = (c < DT_RANK) ? (_Float16)src[r * DT_RANK + c] : (_Float16)0.0f;
}

__global__ void pad_dtr_kernel(const float* __restrict__ xdbl, _Float16* __restrict__ dtr) {
    int i = blockIdx.x * 256 + threadIdx.x;
    if (i >= M_TOK * DT_PAD) return;
    int m = i / DT_PAD, c = i % DT_PAD;
    dtr[i] = (c < DT_RANK) ? (_Float16)xdbl[(size_t)m * NXPROJ + c] : (_Float16)0.0f;
}

// ---------------------------------------------------------------------------
// WMMA GEMM: C[M,N](f32) = A[M,K](f16,rm) * B[N,K](f16,rm)^T (+ residual)
// K % 32 == 0, M % 128 == 0; N guarded only when NGUARD.
// 256 threads = 8 waves (2x4); tile 128x128, BK=32; wave does 4x2 WMMA tiles.
// Async double-buffered LDS staging (ASYNCcnt) when available.
// ---------------------------------------------------------------------------
#define BM 128
#define BN 128
#define BK 32
#define LDS_STRIDE 40   /* halfs; 80 B rows keep every v8h chunk 16B-aligned */

template<bool NGUARD>
__global__ __launch_bounds__(256)
void gemm_kernel(const _Float16* __restrict__ A, const _Float16* __restrict__ Bmat,
                 float* __restrict__ C, const float* __restrict__ resid,
                 int Ndim, int Kdim) {
    __shared__ __align__(16) _Float16 As[2][BM * LDS_STRIDE];
    __shared__ __align__(16) _Float16 Bs[2][BN * LDS_STRIDE];

    const int m0 = blockIdx.y * BM;
    const int n0 = blockIdx.x * BN;
    const int tid  = threadIdx.x;
    const int lane = tid & 31;
    const int wave = tid >> 5;
    const int wy = wave >> 2;        // 0..1 -> 64-row slab
    const int wx = wave & 3;         // 0..3 -> 32-col slab
    const int g = lane >> 4;         // lane group (ISA 7.12.2 fragment layouts)
    const int r = lane & 15;

    v8f acc[4][2];
    #pragma unroll
    for (int mi = 0; mi < 4; mi++)
        #pragma unroll
        for (int ni = 0; ni < 2; ni++)
            #pragma unroll
            for (int e = 0; e < 8; e++) acc[mi][ni][e] = 0.0f;

    const v8h zero8 = {};

#if USE_ASYNC
    // Tile staging via gfx1250 async global->LDS (no VGPR round-trip).
    auto stage = [&](int bi, int k0) {
        #pragma unroll
        for (int i = 0; i < 2; i++) {
            int c   = tid + 256 * i;       // 0..511 chunk id (16B chunks)
            int row = c >> 2;
            int kc  = (c & 3) * 8;
            __builtin_amdgcn_global_load_async_to_lds_b128(
                (AS1 v4i*)(A + (size_t)(m0 + row) * Kdim + k0 + kc),
                (AS3 v4i*)(&As[bi][row * LDS_STRIDE + kc]), 0, 0);
        }
        #pragma unroll
        for (int i = 0; i < 2; i++) {
            int c   = tid + 256 * i;
            int row = c >> 2;
            int kc  = (c & 3) * 8;
            if (!NGUARD || (n0 + row) < Ndim)
                __builtin_amdgcn_global_load_async_to_lds_b128(
                    (AS1 v4i*)(Bmat + (size_t)(n0 + row) * Kdim + k0 + kc),
                    (AS3 v4i*)(&Bs[bi][row * LDS_STRIDE + kc]), 0, 0);
        }
    };

    if (NGUARD) {
        // out-of-range B rows are never async-written: zero them once, both bufs
        #pragma unroll
        for (int i = 0; i < 2; i++) {
            int c   = tid + 256 * i;
            int row = c >> 2;
            int kc  = (c & 3) * 8;
            if (n0 + row >= Ndim) {
                *(v8h*)(&Bs[0][row * LDS_STRIDE + kc]) = zero8;
                *(v8h*)(&Bs[1][row * LDS_STRIDE + kc]) = zero8;
            }
        }
    }

    stage(0, 0);
    const int nIter = Kdim / BK;
    for (int it = 0; it < nIter; ++it) {
        __builtin_amdgcn_s_wait_asynccnt(0);   // own async ops for tile `it` done
        __syncthreads();                       // everyone's tile data visible
        const int cur = it & 1;
        if (it + 1 < nIter) stage(cur ^ 1, (it + 1) * BK);

        Frag16 fa[4];
        #pragma unroll
        for (int mi = 0; mi < 4; mi++) {
            const _Float16* p = &As[cur][(wy * 64 + mi * 16 + r) * LDS_STRIDE];
            fa[mi].h[0] = *(const v8h*)(p + g * 8);
            fa[mi].h[1] = *(const v8h*)(p + 16 + g * 8);
        }
        Frag16 fb[2];
        #pragma unroll
        for (int ni = 0; ni < 2; ni++) {
            const _Float16* p = &Bs[cur][(wx * 32 + ni * 16 + r) * LDS_STRIDE];
            fb[ni].h[0] = *(const v8h*)(p + g * 8);
            fb[ni].h[1] = *(const v8h*)(p + 16 + g * 8);
        }
        #pragma unroll
        for (int mi = 0; mi < 4; mi++)
            #pragma unroll
            for (int ni = 0; ni < 2; ni++)
                acc[mi][ni] = __builtin_amdgcn_wmma_f32_16x16x32_f16(
                    false, fa[mi].v, false, fb[ni].v,
                    (short)0, acc[mi][ni], false, false);
        // ds reads drained before the consuming wmma -> next-iter barrier is a
        // sufficient fence for buffer reuse.
    }
#else
    // Fallback: synchronous staging, single buffer.
    for (int k0 = 0; k0 < Kdim; k0 += BK) {
        #pragma unroll
        for (int i = 0; i < 2; i++) {
            int c   = tid + 256 * i;
            int row = c >> 2;
            int kc  = (c & 3) * 8;
            *(v8h*)(&As[0][row * LDS_STRIDE + kc]) =
                *(const v8h*)(A + (size_t)(m0 + row) * Kdim + k0 + kc);
        }
        #pragma unroll
        for (int i = 0; i < 2; i++) {
            int c   = tid + 256 * i;
            int row = c >> 2;
            int kc  = (c & 3) * 8;
            v8h val = zero8;
            if (!NGUARD || (n0 + row) < Ndim)
                val = *(const v8h*)(Bmat + (size_t)(n0 + row) * Kdim + k0 + kc);
            *(v8h*)(&Bs[0][row * LDS_STRIDE + kc]) = val;
        }
        __syncthreads();
        Frag16 fa[4];
        #pragma unroll
        for (int mi = 0; mi < 4; mi++) {
            const _Float16* p = &As[0][(wy * 64 + mi * 16 + r) * LDS_STRIDE];
            fa[mi].h[0] = *(const v8h*)(p + g * 8);
            fa[mi].h[1] = *(const v8h*)(p + 16 + g * 8);
        }
        Frag16 fb[2];
        #pragma unroll
        for (int ni = 0; ni < 2; ni++) {
            const _Float16* p = &Bs[0][(wx * 32 + ni * 16 + r) * LDS_STRIDE];
            fb[ni].h[0] = *(const v8h*)(p + g * 8);
            fb[ni].h[1] = *(const v8h*)(p + 16 + g * 8);
        }
        #pragma unroll
        for (int mi = 0; mi < 4; mi++)
            #pragma unroll
            for (int ni = 0; ni < 2; ni++)
                acc[mi][ni] = __builtin_amdgcn_wmma_f32_16x16x32_f16(
                    false, fa[mi].v, false, fb[ni].v,
                    (short)0, acc[mi][ni], false, false);
        __syncthreads();
    }
#endif

    // store: D layout row = vgpr + 8*g, col = r (per 16x16 tile)
    #pragma unroll
    for (int mi = 0; mi < 4; mi++) {
        #pragma unroll
        for (int ni = 0; ni < 2; ni++) {
            int col = n0 + wx * 32 + ni * 16 + r;
            if (!NGUARD || col < Ndim) {
                #pragma unroll
                for (int rr = 0; rr < 8; rr++) {
                    int row = m0 + wy * 64 + mi * 16 + rr + 8 * g;
                    size_t idx = (size_t)row * Ndim + col;
                    float v = acc[mi][ni][rr];
                    if (resid) v += resid[idx];
                    C[idx] = v;
                }
            }
        }
    }
}

// ---------------------------------------------------------------------------
// Depthwise causal conv (kernel 4) + bias + SiLU; emits f32 and f16 copies
// ---------------------------------------------------------------------------
__global__ void conv_silu_kernel(const float* __restrict__ xz, const float* __restrict__ cw,
                                 const float* __restrict__ cb, float* __restrict__ xc,
                                 _Float16* __restrict__ xcf) {
    int i = blockIdx.x * 256 + threadIdx.x;   // over M_TOK * D_INNER
    if (i >= M_TOK * D_INNER) return;
    int d = i % D_INNER;
    int tok = i / D_INNER;
    int t = tok % T_LEN;
    float acc = cb[d];
    #pragma unroll
    for (int j = 0; j < 4; j++) {
        int tt = t - 3 + j;
        if (tt >= 0)
            acc += cw[d * 4 + j] * xz[(size_t)(tok - (3 - j)) * D2 + d];
    }
    float sv = acc / (1.0f + __expf(-acc));   // SiLU
    xc[i] = sv;
    xcf[i] = (_Float16)sv;
}

// ---------------------------------------------------------------------------
// Selective scan: one channel per lane, 16 states in registers.
// Fuses softplus(dt), dA/dBu, y = (scan + D*xc) * silu(z)  ->  f16 y.
// B/C rows broadcast through double-buffered LDS (one barrier per step).
// ---------------------------------------------------------------------------
__global__ __launch_bounds__(128)
void scan_kernel(const float* __restrict__ dtpre, const float* __restrict__ bdt,
                 const float* __restrict__ xc, const float* __restrict__ xdbl,
                 const float* __restrict__ Alog, const float* __restrict__ Dp,
                 const float* __restrict__ xz, _Float16* __restrict__ yf) {
    const int bIdx = blockIdx.x / (D_INNER / 128);
    const int dBlk = blockIdx.x % (D_INNER / 128);
    const int d = dBlk * 128 + threadIdx.x;

    __shared__ float BC[2][32];   // [buf][ B:0..15 | C:16..31 ]

    float a[D_STATE], s[D_STATE];
    #pragma unroll
    for (int n = 0; n < D_STATE; n++) {
        a[n] = -__expf(Alog[(size_t)d * D_STATE + n]);
        s[n] = 0.0f;
    }
    const float bias = bdt[d];
    const float dpv  = Dp[d];

    for (int t = 0; t < T_LEN; t++) {
        const int row = bIdx * T_LEN + t;
        const int buf = t & 1;
        if (threadIdx.x < 32)
            BC[buf][threadIdx.x] = xdbl[(size_t)row * NXPROJ + DT_RANK + threadIdx.x];
        __syncthreads();

        float dtv = dtpre[(size_t)row * D_INNER + d] + bias;
        dtv = (dtv > 20.0f) ? dtv : log1pf(__expf(dtv));   // softplus
        const float xcv = xc[(size_t)row * D_INNER + d];
        const float dtx = dtv * xcv;

        float y = 0.0f;
        #pragma unroll
        for (int n = 0; n < D_STATE; n++) {
            float dA = __expf(dtv * a[n]);
            s[n] = dA * s[n] + dtx * BC[buf][n];
            y += s[n] * BC[buf][16 + n];
        }
        const float z = xz[(size_t)row * D2 + D_INNER + d];
        const float yo = (y + dpv * xcv) * (z / (1.0f + __expf(-z)));
        yf[(size_t)row * D_INNER + d] = (_Float16)yo;
    }
}

// ---------------------------------------------------------------------------
extern "C" void kernel_launch(void* const* d_in, const int* in_sizes, int n_in,
                              void* d_out, int out_size, void* d_ws, size_t ws_size,
                              hipStream_t stream) {
    const float* x    = (const float*)d_in[0];
    const float* lng  = (const float*)d_in[1];
    const float* lnb  = (const float*)d_in[2];
    const float* Win  = (const float*)d_in[3];
    const float* cw   = (const float*)d_in[4];
    const float* cb   = (const float*)d_in[5];
    const float* Wx   = (const float*)d_in[6];
    const float* Wdt  = (const float*)d_in[7];
    const float* bdt  = (const float*)d_in[8];
    const float* Alog = (const float*)d_in[9];
    const float* Dp   = (const float*)d_in[10];
    const float* Wout = (const float*)d_in[11];
    float* out = (float*)d_out;

    char* ws = (char*)d_ws;
    size_t off = 0;
    auto alloc = [&](size_t bytes) -> void* {
        void* p = ws + off;
        off = (off + bytes + 255) & ~(size_t)255;
        return p;
    };

    _Float16* h16    = (_Float16*)alloc((size_t)M_TOK * D_MODEL * 2);
    _Float16* Win16  = (_Float16*)alloc((size_t)D2 * D_MODEL * 2);
    _Float16* Wx16   = (_Float16*)alloc((size_t)NXPROJ * D_INNER * 2);
    _Float16* Wdt16  = (_Float16*)alloc((size_t)D_INNER * DT_PAD * 2);
    _Float16* Wout16 = (_Float16*)alloc((size_t)D_MODEL * D_INNER * 2);
    float*    xzbuf  = (float*)   alloc((size_t)M_TOK * D2 * 4);
    float*    xcbuf  = (float*)   alloc((size_t)M_TOK * D_INNER * 4);
    _Float16* xc16   = (_Float16*)alloc((size_t)M_TOK * D_INNER * 2);
    float*    xdbl   = (float*)   alloc((size_t)M_TOK * NXPROJ * 4);
    _Float16* dtr16  = (_Float16*)alloc((size_t)M_TOK * DT_PAD * 2);
    float*    dtpre  = (float*)   alloc((size_t)M_TOK * D_INNER * 4);
    _Float16* y16    = (_Float16*)alloc((size_t)M_TOK * D_INNER * 2);

    // Weight conversions (deterministic every call)
    {
        int n = D2 * D_MODEL;
        cvt_kernel<<<(n + 255) / 256, 256, 0, stream>>>(Win, Win16, n);
        n = NXPROJ * D_INNER;
        cvt_kernel<<<(n + 255) / 256, 256, 0, stream>>>(Wx, Wx16, n);
        n = D_MODEL * D_INNER;
        cvt_kernel<<<(n + 255) / 256, 256, 0, stream>>>(Wout, Wout16, n);
        n = D_INNER * DT_PAD;
        pad_wdt_kernel<<<(n + 255) / 256, 256, 0, stream>>>(Wdt, Wdt16);
    }

    // 1) LayerNorm -> f16 h
    ln_kernel<<<M_TOK, 256, 0, stream>>>(x, lng, lnb, h16);

    // 2) xz = h @ W_in^T   [2048 x 3072], K=768
    gemm_kernel<false><<<dim3(D2 / BN, M_TOK / BM), 256, 0, stream>>>(
        h16, Win16, xzbuf, nullptr, D2, D_MODEL);

    // 3) depthwise causal conv + SiLU
    conv_silu_kernel<<<(M_TOK * D_INNER) / 256, 256, 0, stream>>>(
        xzbuf, cw, cb, xcbuf, xc16);

    // 4) x_dbl = xc @ W_xproj^T   [2048 x 80], K=1536  (N-guarded)
    gemm_kernel<true><<<dim3(1, M_TOK / BM), 256, 0, stream>>>(
        xc16, Wx16, xdbl, nullptr, NXPROJ, D_INNER);

    // 5) pad dt_r to K=64 f16
    pad_dtr_kernel<<<(M_TOK * DT_PAD) / 256, 256, 0, stream>>>(xdbl, dtr16);

    // 6) dt_pre = dt_r @ W_dt^T   [2048 x 1536], K=64
    gemm_kernel<false><<<dim3(D_INNER / BN, M_TOK / BM), 256, 0, stream>>>(
        dtr16, Wdt16, dtpre, nullptr, D_INNER, DT_PAD);

    // 7) selective scan (fused softplus + SSM + D-skip + SiLU gate) -> f16 y
    scan_kernel<<<NB * (D_INNER / 128), 128, 0, stream>>>(
        dtpre, bdt, xcbuf, xdbl, Alog, Dp, xzbuf, y16);

    // 8) out = x + y @ W_out^T   [2048 x 768], K=1536, fused residual
    gemm_kernel<false><<<dim3(D_MODEL / BN, M_TOK / BM), 256, 0, stream>>>(
        y16, Wout16, out, x, D_MODEL, D_INNER);
}
